// ExpansionContrastModule_7352984011497
// MI455X (gfx1250) — matrix operand order repaired and checked by reading, not compiled, not run on hardware
//
#include <hip/hip_runtime.h>
#include <hip/hip_bf16.h>

typedef __attribute__((ext_vector_type(16))) __bf16 v16bf;
typedef __attribute__((ext_vector_type(8)))  float  v8f;
typedef __attribute__((ext_vector_type(4)))  unsigned int u32x4;

union Frag {
    v16bf v;
    u32x4 q[2];
};

static constexpr int BN   = 4;
static constexpr int CIN  = 256;
static constexpr int CC   = 64;
static constexpr int IH   = 160;
static constexpr int IW   = 160;
static constexpr int HW   = IH * IW;        // 25600
static constexpr int NPIX = BN * HW;        // 102400
static constexpr int NELT = BN * CC * HW;   // 6553600

// ---------------- weight fp32 -> bf16 conversion (K-contiguous already) ----------------
__global__ void k_cvt(const float* __restrict__ src, __bf16* __restrict__ dst, int n) {
    int i = blockIdx.x * blockDim.x + threadIdx.x;
    if (i < n) dst[i] = (__bf16)src[i];
}

// ---- conv weight OIHW (CC,CC,KS,KS) -> (CC,KS,KS,CC) bf16, ic contiguous ----
__global__ void k_cvt_tr(const float* __restrict__ src, __bf16* __restrict__ dst, int KS) {
    int i = blockIdx.x * blockDim.x + threadIdx.x;
    int total = CC * CC * KS * KS;
    if (i >= total) return;
    int ic = i % CC;
    int t = i / CC;
    int kx = t % KS; t /= KS;
    int ky = t % KS;
    int oc = t / KS;
    dst[i] = (__bf16)src[((oc * CC + ic) * KS + ky) * KS + kx];
}

// -------- 1x1 conv 256->64 : WMMA GEMM, M-blocked (wave does all 64 oc) --------
__global__ void k_inconv(const float* __restrict__ cen, const __bf16* __restrict__ wbf,
                         const float* __restrict__ bias, __bf16* __restrict__ x0t) {
    int nt = (blockIdx.x * blockDim.x + threadIdx.x) >> 5;  // pixel tile
    int lane = threadIdx.x & 31;
    int pix0 = nt * 16;
    int b = pix0 / HW, hw0 = pix0 % HW;
    int n = lane & 15, khalf = lane >> 4, m = lane & 15;

    v8f acc[4];
#pragma unroll
    for (int mt = 0; mt < 4; ++mt)
#pragma unroll
        for (int r = 0; r < 8; ++r) acc[mt][r] = bias[mt * 16 + r + 8 * khalf];

    for (int kc = 0; kc < CIN; kc += 32) {
        Frag B;
        const float* src = cen + ((size_t)b * CIN + kc + khalf * 16) * HW + hw0 + n;
#pragma unroll
        for (int e = 0; e < 16; ++e)
            B.v[e] = (__bf16)src[(size_t)e * HW];
#pragma unroll
        for (int mt = 0; mt < 4; ++mt) {
            const __bf16* arow = wbf + (size_t)(mt * 16 + m) * CIN + kc;
            Frag A;
            A.q[0] = *(const u32x4*)(arow + khalf * 16);
            A.q[1] = *(const u32x4*)(arow + khalf * 16 + 8);
            acc[mt] = __builtin_amdgcn_wmma_f32_16x16x32_bf16(false, A.v, false, B.v,
                                                              (short)0, acc[mt], false, false);
        }
    }
    // pixel-major store: per mt, 8 contiguous channels -> one b128 store per lane
    __bf16* dst = x0t + (size_t)(b * HW + hw0 + n) * CC + 8 * khalf;
#pragma unroll
    for (int mt = 0; mt < 4; ++mt) {
        union { u32x4 q; __bf16 h[8]; } o;
#pragma unroll
        for (int r = 0; r < 8; ++r) o.h[r] = (__bf16)acc[mt][r];
        *(u32x4*)(dst + mt * 16) = o.q;
    }
}

// -------- kxk conv 64->64 : WMMA GEMM, M-blocked, shared B fragment --------
template <int KS>
__global__ void k_conv(const __bf16* __restrict__ x0t, const __bf16* __restrict__ wtr,
                       const float* __restrict__ bias, float* __restrict__ xo) {
    constexpr int P = KS / 2;
    int nt = (blockIdx.x * blockDim.x + threadIdx.x) >> 5;  // pixel tile
    int lane = threadIdx.x & 31;
    int pix0 = nt * 16;
    int b = pix0 / HW, hw0 = pix0 % HW;
    int h = hw0 / IW, w0 = hw0 % IW;
    int n = lane & 15, khalf = lane >> 4, m = lane & 15;
    int w = w0 + n;

    v8f acc[4];
#pragma unroll
    for (int mt = 0; mt < 4; ++mt)
#pragma unroll
        for (int r = 0; r < 8; ++r) acc[mt][r] = bias[mt * 16 + r + 8 * khalf];

    for (int ky = 0; ky < KS; ++ky) {
        int hh = h + ky - P;
        bool hok = (hh >= 0) & (hh < IH);
        int hhc = hh < 0 ? 0 : (hh >= IH ? IH - 1 : hh);
        // prefetch next halo row (gfx1250 global_prefetch_b8)
        if (ky + 1 < KS) {
            int hp = h + ky + 1 - P;
            hp = hp < 0 ? 0 : (hp >= IH ? IH - 1 : hp);
            __builtin_prefetch(x0t + (size_t)(b * HW + hp * IW + w) * CC, 0, 0);
        }
        for (int kx = 0; kx < KS; ++kx) {
            int ww = w + kx - P;
            bool ok = hok & (ww >= 0) & (ww < IW);
            int wwc = ww < 0 ? 0 : (ww >= IW ? IW - 1 : ww);
            const __bf16* bsrc = x0t + (size_t)(b * HW + hhc * IW + wwc) * CC + khalf * 16;
            const __bf16* asrc = wtr + (size_t)m * (KS * KS * CC) + (ky * KS + kx) * CC;
#pragma unroll
            for (int kc = 0; kc < CC; kc += 32) {
                Frag B;
                B.q[0] = *(const u32x4*)(bsrc + kc);
                B.q[1] = *(const u32x4*)(bsrc + kc + 8);
                if (!ok) {  // zero-pad border: data select, no exec divergence at WMMA
                    u32x4 z = {0u, 0u, 0u, 0u};
                    B.q[0] = z;
                    B.q[1] = z;
                }
#pragma unroll
                for (int mt = 0; mt < 4; ++mt) {
                    const __bf16* ap = asrc + (size_t)(mt * 16) * (KS * KS * CC) + kc;
                    Frag A;
                    A.q[0] = *(const u32x4*)(ap + khalf * 16);
                    A.q[1] = *(const u32x4*)(ap + khalf * 16 + 8);
                    acc[mt] = __builtin_amdgcn_wmma_f32_16x16x32_bf16(false, A.v, false, B.v,
                                                                      (short)0, acc[mt],
                                                                      false, false);
                }
            }
        }
    }
#pragma unroll
    for (int mt = 0; mt < 4; ++mt)
#pragma unroll
        for (int r = 0; r < 8; ++r) {
            int oc = mt * 16 + r + 8 * khalf;
            xo[((size_t)b * CC + oc) * HW + hw0 + n] = acc[mt][r];
        }
}

// ---------------- contrast products + sort4 + scale dot, running max/sum ----------------
__global__ void k_contrast(const float* __restrict__ x, const float* __restrict__ scales,
                           float* __restrict__ ymax, float* __restrict__ ysum, int d,
                           int first) {
    int i = blockIdx.x * blockDim.x + threadIdx.x;
    if (i >= NELT) return;
    int hw = i % HW;
    int bc = i / HW;
    int g = bc % CC;
    int h = hw / IW, w = hw % IW;
    const float* xp = x + (size_t)bc * HW;
    float v = xp[hw];
    auto nb = [&](int dh, int dw) -> float {
        int hh = h + dh, ww = w + dw;
        if (hh < 0 || hh >= IH || ww < 0 || ww >= IW) return 0.0f;
        return xp[hh * IW + ww];
    };
    float o0 = (v - nb(-d, -d)) * (v - nb(d, d));
    float o1 = (v - nb(-d, 0)) * (v - nb(d, 0));
    float o2 = (v - nb(-d, d)) * (v - nb(d, -d));
    float o3 = (v - nb(0, -d)) * (v - nb(0, d));
    float t;
#define CSWAP(a, b)          \
    if (a > b) {             \
        t = a; a = b; b = t; \
    }
    CSWAP(o0, o1) CSWAP(o2, o3) CSWAP(o0, o2) CSWAP(o1, o3) CSWAP(o1, o2)
#undef CSWAP
    float y = o0 * scales[g * 4 + 0] + o1 * scales[g * 4 + 1] + o2 * scales[g * 4 + 2] +
              o3 * scales[g * 4 + 3];
    if (first) {
        ymax[i] = y;
        ysum[i] = y;
    } else {
        ymax[i] = fmaxf(ymax[i], y);
        ysum[i] = ysum[i] + y;
    }
}

// ---------------- m = relu(max + mean) ----------------
__global__ void k_m(const float* __restrict__ ymax, const float* __restrict__ ysum,
                    float* __restrict__ m) {
    int i = blockIdx.x * blockDim.x + threadIdx.x;
    if (i >= NELT) return;
    float v = ymax[i] + ysum[i] * 0.25f;
    m[i] = v > 0.0f ? v : 0.0f;
}

// -------- 1x1 conv 64->64 + BN + SiLU : WMMA GEMM, M-blocked --------
__global__ void k_bc(const float* __restrict__ m, const __bf16* __restrict__ wbf,
                     const float* __restrict__ gma, const float* __restrict__ bta,
                     const float* __restrict__ mu, const float* __restrict__ var,
                     float* __restrict__ z) {
    int nt = (blockIdx.x * blockDim.x + threadIdx.x) >> 5;
    int lane = threadIdx.x & 31;
    int pix0 = nt * 16;
    int b = pix0 / HW, hw0 = pix0 % HW;
    int n = lane & 15, khalf = lane >> 4, mm = lane & 15;

    v8f acc[4];
#pragma unroll
    for (int mt = 0; mt < 4; ++mt)
#pragma unroll
        for (int r = 0; r < 8; ++r) acc[mt][r] = 0.0f;

#pragma unroll
    for (int kc = 0; kc < CC; kc += 32) {
        Frag B;
        const float* src = m + ((size_t)b * CC + kc + khalf * 16) * HW + hw0 + n;
#pragma unroll
        for (int e = 0; e < 16; ++e)
            B.v[e] = (__bf16)src[(size_t)e * HW];
#pragma unroll
        for (int mt = 0; mt < 4; ++mt) {
            const __bf16* arow = wbf + (size_t)(mt * 16 + mm) * CC + kc;
            Frag A;
            A.q[0] = *(const u32x4*)(arow + khalf * 16);
            A.q[1] = *(const u32x4*)(arow + khalf * 16 + 8);
            acc[mt] = __builtin_amdgcn_wmma_f32_16x16x32_bf16(false, A.v, false, B.v,
                                                              (short)0, acc[mt], false, false);
        }
    }
#pragma unroll
    for (int mt = 0; mt < 4; ++mt)
#pragma unroll
        for (int r = 0; r < 8; ++r) {
            int oc = mt * 16 + r + 8 * khalf;
            float s = gma[oc] * rsqrtf(var[oc] + 1e-5f);
            float zz = (acc[mt][r] - mu[oc]) * s + bta[oc];
            zz = zz / (1.0f + __expf(-zz));  // SiLU
            z[((size_t)b * CC + oc) * HW + hw0 + n] = zz;
        }
}

// ---------------- logits -> mask -> out = cen*(mask+1) ----------------
__global__ void k_final(const float* __restrict__ cen, const float* __restrict__ z,
                        const float* __restrict__ fcw, const float* __restrict__ fcb,
                        float* __restrict__ out) {
    int p = blockIdx.x * blockDim.x + threadIdx.x;
    if (p >= NPIX) return;
    int b = p / HW, hw = p % HW;
    float s = fcb[0];
    const float* zp = z + (size_t)b * CC * HW + hw;
#pragma unroll 8
    for (int ch = 0; ch < CC; ++ch) s += fcw[ch] * zp[(size_t)ch * HW];
    float mask = 1.0f / (1.0f + __expf(-s));
    float f = mask + 1.0f;
    const float* cp = cen + (size_t)b * CIN * HW + hw;
    float* op = out + (size_t)b * CIN * HW + hw;
#pragma unroll 4
    for (int ch = 0; ch < CIN; ++ch) op[(size_t)ch * HW] = cp[(size_t)ch * HW] * f;
}

extern "C" void kernel_launch(void* const* d_in, const int* in_sizes, int n_in,
                              void* d_out, int out_size, void* d_ws, size_t ws_size,
                              hipStream_t stream) {
    (void)in_sizes; (void)n_in; (void)out_size; (void)ws_size;

    const float* cen   = (const float*)d_in[0];
    // d_in[1] = mas : unused by the reference
    const float* inw   = (const float*)d_in[2];
    const float* inb   = (const float*)d_in[3];
    const float* cw[4] = {(const float*)d_in[4], (const float*)d_in[6],
                          (const float*)d_in[8], (const float*)d_in[10]};
    const float* cb[4] = {(const float*)d_in[5], (const float*)d_in[7],
                          (const float*)d_in[9], (const float*)d_in[11]};
    const float* scales = (const float*)d_in[12];
    const float* bcw    = (const float*)d_in[13];
    const float* gma    = (const float*)d_in[14];
    const float* bta    = (const float*)d_in[15];
    const float* mu     = (const float*)d_in[16];
    const float* var    = (const float*)d_in[17];
    const float* fcw    = (const float*)d_in[18];
    const float* fcb    = (const float*)d_in[19];
    float* out = (float*)d_out;

    char* ws = (char*)d_ws;
    size_t OFF_X0  = 0;                                   // bf16 x0 (pixel-major)
    size_t OFF_X   = OFF_X0 + (size_t)NELT * 2;           // f32 x / m (NCHW)
    size_t OFF_YMX = OFF_X + (size_t)NELT * 4;            // f32 ymax / z
    size_t OFF_YSM = OFF_YMX + (size_t)NELT * 4;          // f32 ysum
    size_t OFF_WIN = OFF_YSM + (size_t)NELT * 4;          // bf16 weights below
    size_t OFF_W1  = OFF_WIN + (size_t)CC * CIN * 2;
    size_t OFF_W3  = OFF_W1 + (size_t)CC * CC * 2;
    size_t OFF_W5  = OFF_W3 + (size_t)CC * CC * 9 * 2;
    size_t OFF_W7  = OFF_W5 + (size_t)CC * CC * 25 * 2;
    size_t OFF_WBC = OFF_W7 + (size_t)CC * CC * 49 * 2;

    __bf16* x0t  = (__bf16*)(ws + OFF_X0);
    float*  xbuf = (float*)(ws + OFF_X);
    float*  ymax = (float*)(ws + OFF_YMX);
    float*  ysum = (float*)(ws + OFF_YSM);
    __bf16* winb = (__bf16*)(ws + OFF_WIN);
    __bf16* wkb[4] = {(__bf16*)(ws + OFF_W1), (__bf16*)(ws + OFF_W3),
                      (__bf16*)(ws + OFF_W5), (__bf16*)(ws + OFF_W7)};
    __bf16* wbcb = (__bf16*)(ws + OFF_WBC);

    // weight conversions (+ transpose for kxk convs: OIHW -> O,ky,kx,I)
    k_cvt<<<(CC * CIN + 255) / 256, 256, 0, stream>>>(inw, winb, CC * CIN);
    int ksz[4] = {1, 3, 5, 7};
    for (int i = 0; i < 4; ++i) {
        int wn = CC * CC * ksz[i] * ksz[i];
        k_cvt_tr<<<(wn + 255) / 256, 256, 0, stream>>>(cw[i], wkb[i], ksz[i]);
    }
    k_cvt<<<(CC * CC + 255) / 256, 256, 0, stream>>>(bcw, wbcb, CC * CC);

    // GEMM grid: one wave per 16-pixel tile (wave covers all 64 oc), 4 waves/block
    dim3 gemmGrid(NPIX / 16 / 4), gemmBlk(128);  // 1600 blocks

    // stage 1: 1x1 conv 256->64 (WMMA), bf16 pixel-major activations
    k_inconv<<<gemmGrid, gemmBlk, 0, stream>>>(cen, winb, inb, x0t);

    // stage 2: four branches — conv k (WMMA) then contrast/sort/scale
    dim3 eltGrid((NELT + 255) / 256), eltBlk(256);
    int shifts[4] = {1, 3, 5, 7};
    for (int br = 0; br < 4; ++br) {
        switch (shifts[br]) {
            case 1: k_conv<1><<<gemmGrid, gemmBlk, 0, stream>>>(x0t, wkb[br], cb[br], xbuf); break;
            case 3: k_conv<3><<<gemmGrid, gemmBlk, 0, stream>>>(x0t, wkb[br], cb[br], xbuf); break;
            case 5: k_conv<5><<<gemmGrid, gemmBlk, 0, stream>>>(x0t, wkb[br], cb[br], xbuf); break;
            case 7: k_conv<7><<<gemmGrid, gemmBlk, 0, stream>>>(x0t, wkb[br], cb[br], xbuf); break;
        }
        k_contrast<<<eltGrid, eltBlk, 0, stream>>>(xbuf, scales, ymax, ysum, shifts[br],
                                                   br == 0 ? 1 : 0);
    }

    // stage 3: m = relu(max + mean)  (into xbuf, x no longer needed)
    k_m<<<eltGrid, eltBlk, 0, stream>>>(ymax, ysum, xbuf);

    // stage 4: 1x1 conv + BN + SiLU (WMMA), z into ymax buffer
    k_bc<<<gemmGrid, gemmBlk, 0, stream>>>(xbuf, wbcb, gma, bta, mu, var, ymax);

    // stage 5: per-pixel logits -> mask -> scale cen
    k_final<<<(NPIX + 255) / 256, 256, 0, stream>>>(cen, ymax, fcw, fcb, out);
}
